// ISOmapModel_20066087206965
// MI455X (gfx1250) — matrix operand order repaired and checked by reading, not compiled, not run on hardware
//
#include <hip/hip_runtime.h>
#include <hip/hip_bf16.h>

#define N_PTS 768
#define N_FEA 256
#define KNN 5
#define FWT 64
#define NT (N_PTS / FWT)

typedef __attribute__((ext_vector_type(16))) _Float16 v16h;
typedef __attribute__((ext_vector_type(8)))  _Float16 v8h;
typedef __attribute__((ext_vector_type(8)))  float    v8f;

// ---------------------------------------------------------------------------
// f32 -> f16 conversion
// ---------------------------------------------------------------------------
__global__ void f32_to_f16_k(const float* __restrict__ src, _Float16* __restrict__ dst, int total) {
    int i = blockIdx.x * blockDim.x + threadIdx.x;
    if (i < total) dst[i] = (_Float16)src[i];
}

// ---------------------------------------------------------------------------
// G = A * A^T via v_wmma_f32_16x16x32_f16. One wave (32 lanes) per 16x16 tile.
// A is row-major [n][kdim] f16, kdim % 32 == 0. Output f32 row-major [n][n].
//
// A-fragment (16x32): lanes 0-15 row m=lane, halves 0..15 = K 0..15 contiguous;
//   lanes 16-31 same rows, halves 0..15 = K 16..31.
// B-fragment (32x16): lane holds column n=lane&15; lanes 0-15 halves 0..15 =
//   K 0..15 contiguous; lanes 16-31 = K 16..31 (same order as A).
// For A*A^T, B(k,n) = A[tile_j*16+n][k], so both fragments read rows of A.
// ---------------------------------------------------------------------------
__global__ void wmma_aat_k(const _Float16* __restrict__ A, float* __restrict__ G,
                           int n, int kdim) {
    const int ti = blockIdx.y, tj = blockIdx.x;
    const int lane = threadIdx.x;              // 0..31
    const int m = lane & 15;
    const int ahalf = (lane >> 4) * 8;         // 0 or 8
    const int bhalf = (lane >> 4) * 16;        // 0 or 16
    const _Float16* rowA = A + (size_t)(ti * 16 + m) * kdim;
    const _Float16* rowB = A + (size_t)(tj * 16 + m) * kdim;

    v8f acc = {};
    for (int kb = 0; kb < kdim; kb += 32) {
        v16h afrag, bfrag;
        afrag = *reinterpret_cast<const v16h*>(rowA + kb + bhalf);
        bfrag = *reinterpret_cast<const v16h*>(rowB + kb + bhalf);
        acc = __builtin_amdgcn_wmma_f32_16x16x32_f16(
            false, afrag, false, bfrag, (short)0, acc, false, false);
    }
    // C/D layout: VGPR r -> row r + (lane<16 ? 0 : 8), col = lane & 15
    const int col = lane & 15;
    const int rbase = (lane >> 4) * 8;
    for (int r = 0; r < 8; ++r)
        G[(size_t)(ti * 16 + rbase + r) * n + tj * 16 + col] = acc[r];
}

// ---------------------------------------------------------------------------
// D[i][j] = grad-safe sqrt(max(G[i][i] + G[j][j] - 2 G[i][j], 0))
// ---------------------------------------------------------------------------
__global__ void dist_k(const float* __restrict__ G, float* __restrict__ D, int n) {
    int j = blockIdx.x * blockDim.x + threadIdx.x;
    int i = blockIdx.y;
    if (j >= n) return;
    float d2 = G[(size_t)i * n + i] + G[(size_t)j * n + j] - 2.0f * G[(size_t)i * n + j];
    d2 = fmaxf(d2, 0.0f);
    D[(size_t)i * n + j] = (d2 > 0.0f) ? sqrtf(d2) : 0.0f;
}

__global__ void fill_k(float* __restrict__ p, float val, int total) {
    int i = blockIdx.x * blockDim.x + threadIdx.x;
    if (i < total) p[i] = val;
}

// ---------------------------------------------------------------------------
// kNN adjacency: one thread per row; 5 argmin passes (ties -> lowest index,
// matching top_k). Self (d=0) is included, like sklearn kneighbors.
// ---------------------------------------------------------------------------
__global__ void knn_k(const float* __restrict__ D, float* __restrict__ adj, int n) {
    int i = blockIdx.x * blockDim.x + threadIdx.x;
    if (i >= n) return;
    int chosen[KNN];
    const float* row = D + (size_t)i * n;
    for (int c = 0; c < KNN; ++c) {
        float best = __builtin_inff();
        int bidx = 0;
        for (int j = 0; j < n; ++j) {
            bool used = false;
            for (int c2 = 0; c2 < c; ++c2) used |= (chosen[c2] == j);
            float d = row[j];
            if (!used && d < best) { best = d; bidx = j; }
        }
        chosen[c] = bidx;
        adj[(size_t)i * n + bidx] = best;
    }
}

// ---------------------------------------------------------------------------
// Blocked Floyd-Warshall, tile FWT=64, block = (64,16), 4 rows per thread.
// ---------------------------------------------------------------------------
__global__ void fw_phase1_k(float* __restrict__ sp, int n, int p) {
    __shared__ float S[FWT][FWT + 1];
    const int tx = threadIdx.x, ty = threadIdx.y, pb = p * FWT;
    for (int q = 0; q < 4; ++q) {
        int r = ty + 16 * q;
        S[r][tx] = sp[(size_t)(pb + r) * n + pb + tx];
    }
    __syncthreads();
    for (int k = 0; k < FWT; ++k) {
        float cand[4];
        for (int q = 0; q < 4; ++q) cand[q] = S[ty + 16 * q][k] + S[k][tx];
        __syncthreads();
        for (int q = 0; q < 4; ++q) {
            int i = ty + 16 * q;
            if (cand[q] < S[i][tx]) S[i][tx] = cand[q];
        }
        __syncthreads();
    }
    for (int q = 0; q < 4; ++q) {
        int r = ty + 16 * q;
        sp[(size_t)(pb + r) * n + pb + tx] = S[r][tx];
    }
}

__global__ void fw_phase2_k(float* __restrict__ sp, int n, int p) {
    __shared__ float P[FWT][FWT + 1];
    __shared__ float C[FWT][FWT + 1];
    const int t = blockIdx.x;
    if (t == p) return;
    const bool rowcase = (blockIdx.y == 0);
    const int tx = threadIdx.x, ty = threadIdx.y, pb = p * FWT;
    const int rb = rowcase ? pb : t * FWT;
    const int cb = rowcase ? t * FWT : pb;
    for (int q = 0; q < 4; ++q) {
        int r = ty + 16 * q;
        P[r][tx] = sp[(size_t)(pb + r) * n + pb + tx];
        C[r][tx] = sp[(size_t)(rb + r) * n + cb + tx];
    }
    __syncthreads();
    for (int k = 0; k < FWT; ++k) {
        float cand[4];
        for (int q = 0; q < 4; ++q) {
            int i = ty + 16 * q;
            cand[q] = rowcase ? (P[i][k] + C[k][tx]) : (C[i][k] + P[k][tx]);
        }
        __syncthreads();
        for (int q = 0; q < 4; ++q) {
            int i = ty + 16 * q;
            if (cand[q] < C[i][tx]) C[i][tx] = cand[q];
        }
        __syncthreads();
    }
    for (int q = 0; q < 4; ++q) {
        int r = ty + 16 * q;
        sp[(size_t)(rb + r) * n + cb + tx] = C[r][tx];
    }
}

__global__ void fw_phase3_k(float* __restrict__ sp, int n, int p) {
    __shared__ float Atile[FWT][FWT + 1];
    __shared__ float Btile[FWT][FWT + 1];
    const int r = blockIdx.y, c = blockIdx.x;
    if (r == p || c == p) return;
    const int tx = threadIdx.x, ty = threadIdx.y;
    const int pb = p * FWT, rb = r * FWT, cb = c * FWT;
    for (int q = 0; q < 4; ++q) {
        int i = ty + 16 * q;
        Atile[i][tx] = sp[(size_t)(rb + i) * n + pb + tx];
        Btile[i][tx] = sp[(size_t)(pb + i) * n + cb + tx];
    }
    __syncthreads();
    float cur[4];
    for (int q = 0; q < 4; ++q) cur[q] = sp[(size_t)(rb + ty + 16 * q) * n + cb + tx];
    for (int k = 0; k < FWT; ++k)
        for (int q = 0; q < 4; ++q) {
            float v = Atile[ty + 16 * q][k] + Btile[k][tx];
            if (v < cur[q]) cur[q] = v;
        }
    for (int q = 0; q < 4; ++q) sp[(size_t)(rb + ty + 16 * q) * n + cb + tx] = cur[q];
}

// ---------------------------------------------------------------------------
// max over finite entries (all values >= 0 -> uint-bits atomicMax is valid)
// ---------------------------------------------------------------------------
__global__ void init_scalar_k(unsigned int* s) { *s = 0u; }

__global__ void max_finite_k(const float* __restrict__ sp, unsigned int* s, int total) {
    int i = blockIdx.x * blockDim.x + threadIdx.x;
    if (i < total) {
        float v = sp[i];
        if (v < __builtin_inff()) atomicMax(s, __float_as_uint(v));
    }
}

__global__ void replace_inf_k(float* __restrict__ sp, const unsigned int* s, int total) {
    int i = blockIdx.x * blockDim.x + threadIdx.x;
    if (i < total) {
        float v = sp[i];
        if (!(v < __builtin_inff())) sp[i] = __uint_as_float(*s);
    }
}

// ---------------------------------------------------------------------------
// Double-centering of S = sp*sp: row/col sums of squares, grand total, build B
// ---------------------------------------------------------------------------
__global__ void rowsum_sq_k(const float* __restrict__ sp, float* __restrict__ rs, int n) {
    int i = blockIdx.x * blockDim.x + threadIdx.x;
    if (i >= n) return;
    const float* row = sp + (size_t)i * n;
    float acc = 0.0f;
    for (int j = 0; j < n; ++j) { float v = row[j]; acc = fmaf(v, v, acc); }
    rs[i] = acc;
}

__global__ void colsum_sq_k(const float* __restrict__ sp, float* __restrict__ cs, int n) {
    int j = blockIdx.x * blockDim.x + threadIdx.x;
    if (j >= n) return;
    float acc = 0.0f;
    for (int i = 0; i < n; ++i) { float v = sp[(size_t)i * n + j]; acc = fmaf(v, v, acc); }
    cs[j] = acc;
}

__global__ void totsum_k(const float* __restrict__ rs, float* __restrict__ tot, int n) {
    __shared__ float sh[256];
    float a = 0.0f;
    for (int i = threadIdx.x; i < n; i += 256) a += rs[i];
    sh[threadIdx.x] = a;
    __syncthreads();
    for (int s = 128; s > 0; s >>= 1) {
        if (threadIdx.x < s) sh[threadIdx.x] += sh[threadIdx.x + s];
        __syncthreads();
    }
    if (threadIdx.x == 0) *tot = sh[0];
}

__global__ void build_B_k(const float* __restrict__ sp, const float* __restrict__ rs,
                          const float* __restrict__ cs, const float* __restrict__ tot,
                          float* __restrict__ B, int n) {
    int j = blockIdx.x * blockDim.x + threadIdx.x;
    int i = blockIdx.y;
    if (j >= n) return;
    float fn = (float)n;
    float v = sp[(size_t)i * n + j];
    float s2 = v * v;
    float b = -0.5f * (s2 - rs[i] / fn - cs[j] / fn + (*tot) / (fn * fn));
    B[(size_t)i * n + j] = b;
}

// ---------------------------------------------------------------------------
// Top-2 eigenpairs of M = B B^T by single-workgroup subspace iteration.
// Writes out[i*2+c] = U[i][c] * sqrt(sigma_c), sigma_c = sqrt(lambda_c(M)).
// ---------------------------------------------------------------------------
__device__ float block_reduce_768(float val, float* red, int t) {
    red[t] = val;
    __syncthreads();
    for (int s = 512; s > 0; s >>= 1) {
        if (t < s && t + s < N_PTS) red[t] += red[t + s];
        __syncthreads();
    }
    float r = red[0];
    __syncthreads();
    return r;
}

__global__ void topk_eig_k(const float* __restrict__ M, float* __restrict__ out,
                           int n, int iters) {
    __shared__ float v1[N_PTS], v2[N_PTS], red[N_PTS];
    const int t = threadIdx.x;
    unsigned int h1 = (unsigned)t * 2654435761u + 0x9E3779B9u;
    h1 ^= h1 >> 16; h1 *= 0x85EBCA6Bu; h1 ^= h1 >> 13;
    unsigned int h2 = (unsigned)t * 40503u + 0x27220A95u;
    h2 ^= h2 >> 16; h2 *= 0xC2B2AE35u; h2 ^= h2 >> 13;
    v1[t] = (float)(h1 & 0xFFFFu) / 65536.0f - 0.5f;
    v2[t] = (float)(h2 & 0xFFFFu) / 65536.0f - 0.5f;
    __syncthreads();

    const float* row = M + (size_t)t * n;
    for (int it = 0; it < iters; ++it) {
        float a = 0.0f, b = 0.0f;
        for (int j = 0; j < n; ++j) { float m = row[j]; a = fmaf(m, v1[j], a); b = fmaf(m, v2[j], b); }
        __syncthreads();                         // matvec reads complete
        float n1 = block_reduce_768(a * a, red, t);
        float inv1 = rsqrtf(fmaxf(n1, 1e-30f));
        float u1 = a * inv1;
        v1[t] = u1;
        float d12 = block_reduce_768(u1 * b, red, t);
        float w = b - d12 * u1;
        float n2 = block_reduce_768(w * w, red, t);
        v2[t] = w * rsqrtf(fmaxf(n2, 1e-30f));
        __syncthreads();
    }
    // Rayleigh quotients -> eigenvalues of M -> singular values of B
    float a = 0.0f, b = 0.0f;
    for (int j = 0; j < n; ++j) { float m = row[j]; a = fmaf(m, v1[j], a); b = fmaf(m, v2[j], b); }
    __syncthreads();
    float l1 = block_reduce_768(v1[t] * a, red, t);
    float l2 = block_reduce_768(v2[t] * b, red, t);
    float s1 = sqrtf(fmaxf(l1, 0.0f));
    float s2 = sqrtf(fmaxf(l2, 0.0f));
    out[t * 2 + 0] = v1[t] * sqrtf(s1);
    out[t * 2 + 1] = v2[t] * sqrtf(s2);
}

// ---------------------------------------------------------------------------
extern "C" void kernel_launch(void* const* d_in, const int* in_sizes, int n_in,
                              void* d_out, int out_size, void* d_ws, size_t ws_size,
                              hipStream_t stream) {
    const int n = N_PTS, F = N_FEA;
    const float* X = (const float*)d_in[0];
    float* out = (float*)d_out;
    char* ws = (char*)d_ws;

    const size_t mat = (size_t)n * n * sizeof(float);   // 2,359,296 B
    float*        adj = (float*)(ws + 0 * mat);         // adjacency / geodesics (in place)
    float*        G   = (float*)(ws + 1 * mat);         // Gram, later M = B B^T
    float*        D   = (float*)(ws + 2 * mat);         // distances, later Bh (f16)
    float*        Bm  = (float*)(ws + 3 * mat);         // centered matrix B
    _Float16*     Xh  = (_Float16*)(ws + 4 * mat);      // 768x256 f16 (384 KB)
    float*        rs  = (float*)(ws + 4 * mat + 512 * 1024);
    float*        cs  = rs + n;
    float*        tot = cs + n;
    unsigned int* dmx = (unsigned int*)(tot + 1);
    _Float16*     Bh  = (_Float16*)D;                   // reuse D region

    const int total = n * n;
    const int nb = (total + 255) / 256;

    // 1. X -> f16, Gram via WMMA, distances
    f32_to_f16_k<<<(n * F + 255) / 256, 256, 0, stream>>>(X, Xh, n * F);
    wmma_aat_k<<<dim3(n / 16, n / 16), 32, 0, stream>>>(Xh, G, n, F);
    dist_k<<<dim3(3, n), 256, 0, stream>>>(G, D, n);

    // 2. kNN adjacency (inf elsewhere)
    fill_k<<<nb, 256, 0, stream>>>(adj, __builtin_inff(), total);
    knn_k<<<3, 256, 0, stream>>>(D, adj, n);

    // 3. Blocked Floyd-Warshall APSP
    for (int p = 0; p < NT; ++p) {
        fw_phase1_k<<<1, dim3(FWT, 16), 0, stream>>>(adj, n, p);
        fw_phase2_k<<<dim3(NT, 2), dim3(FWT, 16), 0, stream>>>(adj, n, p);
        fw_phase3_k<<<dim3(NT, NT), dim3(FWT, 16), 0, stream>>>(adj, n, p);
    }

    // 4. Replace unreachable (inf) with max finite geodesic
    init_scalar_k<<<1, 1, 0, stream>>>(dmx);
    max_finite_k<<<nb, 256, 0, stream>>>(adj, dmx, total);
    replace_inf_k<<<nb, 256, 0, stream>>>(adj, dmx, total);

    // 5. Double-centered B = -0.5 * H (sp*sp) H
    rowsum_sq_k<<<3, 256, 0, stream>>>(adj, rs, n);
    colsum_sq_k<<<3, 256, 0, stream>>>(adj, cs, n);
    totsum_k<<<1, 256, 0, stream>>>(rs, tot, n);
    build_B_k<<<dim3(3, n), 256, 0, stream>>>(adj, rs, cs, tot, Bm, n);

    // 6. M = B B^T via WMMA, then top-2 subspace iteration -> embedding
    f32_to_f16_k<<<nb, 256, 0, stream>>>(Bm, Bh, total);
    wmma_aat_k<<<dim3(n / 16, n / 16), 32, 0, stream>>>(Bh, G, n, n);
    topk_eig_k<<<1, n, 0, stream>>>(G, out, n, 300);
}
